// QGCost_and_bilin_60945585930616
// MI455X (gfx1250) — matrix operand order repaired and checked by reading, not compiled, not run on hardware
//
#include <hip/hip_runtime.h>

// ---------- types / helpers ----------
typedef __attribute__((ext_vector_type(16))) __bf16 v16bf;
typedef __attribute__((ext_vector_type(8)))  float  v8f;
typedef unsigned short ushort_t;

union FragAB { v16bf v; uint4 q[2]; };

#define IMG   65536            // 256*256
#define NF    56               // 4 * 14 forecast frames
#define GF    98100.0f         // G/F0
#define KQC   (9.81f * 1.0e-4f / (2.7f * 2.7f))   // G*F0/C0^2
#define PI_F  3.14159265358979f
#define LDA   40               // padded LDS row stride (ushorts): conflict-free b128 frag loads

__device__ __forceinline__ ushort_t f2bf(float f) {
  union { float f; unsigned u; } c; c.f = f;
  unsigned u = c.u;
  unsigned r = (u + 0x7FFFu + ((u >> 16) & 1u)) >> 16;
  return (ushort_t)r;
}

// ---------- init: DX and accumulators ----------
__global__ void k_init0(float* consts) {
  __shared__ float s[256];
  int t = threadIdx.x;
  float lat = 32.0f + 0.05f * (float)t;
  s[t] = __cosf(lat * PI_F / 180.0f);
  __syncthreads();
  for (int o = 128; o > 0; o >>= 1) { if (t < o) s[t] += s[t + o]; __syncthreads(); }
  if (t == 0) {
    float meanc = s[0] / 256.0f;
    float dxm = 0.05f * 111000.0f;
    consts[0] = 0.5f * (dxm + dxm * meanc);   // DX
    consts[8] = 0.0f;                          // dyn sum
    consts[9] = 0.0f;                          // ae sum
  }
}

// ---------- init: DST-I matrix (bf16, padded 256x256) and 1/HELM ----------
__global__ void k_tables(const float* consts, ushort_t* Sb, float* Hinv) {
  int i = blockIdx.x, j = threadIdx.x;
  float DX = consts[0];
  float sv = 0.0f, hv = 1.0f;
  if (i < 254 && j < 254) {
    int m = ((i + 1) * (j + 1)) % 510;         // sin period reduction
    sv = sqrtf(2.0f / 255.0f) * __sinf(PI_F * (float)m / 255.0f);
    float lap = (2.0f * (__cosf(PI_F * (float)(i + 1) / 255.0f) - 1.0f) +
                 2.0f * (__cosf(PI_F * (float)(j + 1) / 255.0f) - 1.0f)) / (DX * DX);
    hv = 1.0f / (GF * lap - KQC);
  }
  Sb[i * 256 + j] = f2bf(sv);
  Hinv[i * 256 + j] = hv;
}

// ---------- copy h0 = x[:, :-1] ----------
__global__ void k_copyh0(const float* xin, float* h) {
  int i = blockIdx.x * 256 + threadIdx.x;
  int f = i >> 16, p = i & 65535;
  int b = f / 14, t = f % 14;
  h[i] = xin[(size_t)(b * 15 + t) * IMG + p];
}

// ---------- qb = h2pv(hb, hb), also seed q ----------
__global__ void k_h2pv(const float* xin, const float* consts, float* qb, float* q) {
  int i = blockIdx.x * 256 + threadIdx.x;
  int f = i >> 16, p = i & 65535;
  int y = p >> 8, x = p & 255;
  int b = f / 14, t = f % 14;
  const float* hb = xin + (size_t)(b * 15 + t) * IMG;
  float DX = consts[0];
  float idx2 = 1.0f / (DX * DX);
  bool mask = (y < 2) || (y > 252) || (x < 2) || (x > 252);
  float hc = hb[p];
  float val;
  if (mask) {
    val = -KQC * hc;
  } else {
    val = GF * ((hb[p + 256] + hb[p - 256] - 2.0f * hc) * idx2 +
                (hb[p + 1]   + hb[p - 1]   - 2.0f * hc) * idx2) - KQC * hc;
  }
  qb[i] = val; q[i] = val;
}

// ---------- q1 = q + DT * rhs(u(h), v(h), q) ----------
__global__ void k_qstep(const float* h, const float* q, float* q1, const float* consts) {
  int i = blockIdx.x * 256 + threadIdx.x;
  int f = i >> 16, p = i & 65535;
  int y = p >> 8, x = p & 255;
  float qc = q[i];
  float out = qc;
  bool mask = (y < 2) || (y > 252) || (x < 2) || (x > 252);
  if (!mask) {
    float DX = consts[0];
    const float* H = h + (size_t)f * IMG;
    const float* Q = q + (size_t)f * IMG;
    float cu = GF * 0.25f / DX;
    float u0 = -cu * (H[(y + 1) * 256 + x - 1] + H[(y + 1) * 256 + x] - H[(y - 1) * 256 + x] - H[(y - 1) * 256 + x - 1]);
    float u1 = -cu * (H[(y + 1) * 256 + x]     + H[(y + 1) * 256 + x + 1] - H[(y - 1) * 256 + x + 1] - H[(y - 1) * 256 + x]);
    float v0 =  cu * (H[y * 256 + x + 1] + H[(y - 1) * 256 + x + 1] - H[(y - 1) * 256 + x - 1] - H[y * 256 + x - 1]);
    float v1 =  cu * (H[(y + 1) * 256 + x + 1] + H[y * 256 + x + 1] - H[y * 256 + x - 1] - H[(y + 1) * 256 + x - 1]);
    float uT = 0.5f * (u0 + u1), vT = 0.5f * (v0 + v1);
    float up = fmaxf(uT, 0.0f), um = fminf(uT, 0.0f);
    float vp = fmaxf(vT, 0.0f), vm = fminf(vT, 0.0f);
    float c = 1.0f / (6.0f * DX);
    float r = -up * c * (2.0f * Q[y * 256 + x + 1] + 3.0f * qc - 6.0f * Q[y * 256 + x - 1] + Q[y * 256 + x - 2])
              + um * c * (Q[y * 256 + x + 2] - 6.0f * Q[y * 256 + x + 1] + 3.0f * qc + 2.0f * Q[y * 256 + x - 1])
              - vp * c * (2.0f * Q[(y + 1) * 256 + x] + 3.0f * qc - 6.0f * Q[(y - 1) * 256 + x] + Q[(y - 2) * 256 + x])
              + vm * c * (Q[(y + 2) * 256 + x] - 6.0f * Q[(y + 1) * 256 + x] + 3.0f * qc + 2.0f * Q[(y - 1) * 256 + x]);
    out = qc + 21600.0f * r;
  }
  q1[i] = out;
}

// ---------- t0b = bf16(embed(q1 - qb interior)), zero-padded ----------
__global__ void k_makeqin(const float* q1, const float* qb, ushort_t* t0b) {
  int i = blockIdx.x * 256 + threadIdx.x;
  int f = i >> 16, p = i & 65535;
  int a = p >> 8, b = p & 255;
  float v = 0.0f;
  if (a < 254 && b < 254) {
    int src = (a + 1) * 256 + (b + 1);
    v = q1[(size_t)f * IMG + src] - qb[(size_t)f * IMG + src];
  }
  t0b[i] = f2bf(v);
}

// ---------- bf16(t * 1/HELM) ----------
__global__ void k_helmcvt(const float* t, const float* Hinv, ushort_t* o) {
  int i = blockIdx.x * 256 + threadIdx.x;
  o[i] = f2bf(t[i] * Hinv[i & 65535]);
}

// ---------- h = embed(inv) + hb ----------
__global__ void k_finish(const float* t0, const float* xin, float* h) {
  int i = blockIdx.x * 256 + threadIdx.x;
  int f = i >> 16, p = i & 65535;
  int y = p >> 8, x = p & 255;
  int b = f / 14, t = f % 14;
  float hb = xin[(size_t)(b * 15 + t) * IMG + p];
  float inner = 0.0f;
  if (y >= 1 && y <= 254 && x >= 1 && x <= 254)
    inner = t0[(size_t)f * IMG + (y - 1) * 256 + (x - 1)];
  h[i] = inner + hb;
}

// ---------- generic f32 -> bf16 convert ----------
__global__ void k_cvt(const float* a, ushort_t* b, int n) {
  int i = blockIdx.x * 256 + threadIdx.x;
  if (i < n) b[i] = f2bf(a[i]);
}

// ---------- bf16(a * b) elementwise ----------
__global__ void k_mulcvt(const float* a, const float* b, ushort_t* o, int n) {
  int i = blockIdx.x * 256 + threadIdx.x;
  if (i < n) o[i] = f2bf(a[i] * b[i]);
}

// ---------- batched 256x256x256 GEMM, bf16 in, f32 accumulate, f32/bf16 out ----------
__global__ __launch_bounds__(256) void k_gemm256(const ushort_t* A, long long sA,
                                                 const ushort_t* B, long long sB,
                                                 void* C, long long sC, int outBf) {
  __shared__ __align__(16) ushort_t As[64 * LDA];
  __shared__ __align__(16) ushort_t Bs[128 * LDA];
  const int tid = threadIdx.x;
  const int wave = tid >> 5, lane = tid & 31;
  const int batch = blockIdx.y;
  const int mb = (blockIdx.x & 3) * 64;
  const int nb = (blockIdx.x >> 2) * 128;
  const ushort_t* Ab = A + sA * batch;
  const ushort_t* Bb = B + sB * batch;
  const int lh = (lane < 16) ? 0 : 1;
  const int ln = lane & 15;

  // this thread's fixed load slots
  const int am = tid >> 2;                 // A row (0..63)
  const int ak = (tid & 3) * 8;            // A k-chunk
  v8f acc[4];
#pragma unroll
  for (int m = 0; m < 4; m++)
#pragma unroll
    for (int r = 0; r < 8; r++) acc[m][r] = 0.0f;

  for (int k0 = 0; k0 < 256; k0 += 32) {
    // A tile 64x32, vectorized copy (global b128 -> LDS b128)
    uint4 va = *(const uint4*)&Ab[(mb + am) * 256 + k0 + ak];
    *(uint4*)&As[am * LDA + ak] = va;
    // B tile 32x128 -> transposed Bs[n][k]
#pragma unroll
    for (int g = 0; g < 2; ++g) {
      int e = tid * 16 + g * 8;
      int k = e >> 7, n = e & 127;
      uint4 vb = *(const uint4*)&Bb[(k0 + k) * 256 + nb + n];
      const ushort_t* pv = (const ushort_t*)&vb;
#pragma unroll
      for (int j = 0; j < 8; j++) Bs[(n + j) * LDA + k] = pv[j];
    }
    if (k0 < 224) {   // prefetch next K tiles into cache (global_prefetch_b8)
      __builtin_prefetch((const void*)&Ab[(mb + am) * 256 + k0 + 32 + ak], 0, 2);
      __builtin_prefetch((const void*)&Bb[(k0 + 32 + (tid >> 4)) * 256 + nb + (tid & 15) * 8], 0, 2);
    }
    __syncthreads();
    FragAB fb;
    {
      int c = wave * 16 + ln;
      int kb = lh ? 8 : 0;
      fb.q[0] = *(const uint4*)&Bs[c * LDA + kb];
      fb.q[1] = *(const uint4*)&Bs[c * LDA + kb + 16];
    }
#pragma unroll
    for (int m = 0; m < 4; m++) {
      FragAB fa;
      int row = m * 16 + ln;
      int ka = lh ? 8 : 0;
      fa.q[0] = *(const uint4*)&As[row * LDA + ka];
      fa.q[1] = *(const uint4*)&As[row * LDA + ka + 16];
      acc[m] = __builtin_amdgcn_wmma_f32_16x16x32_bf16(false, fa.v, false, fb.v,
                                                       (short)0, acc[m], false, false);
    }
    __syncthreads();
  }
  int col = nb + wave * 16 + ln;
  if (outBf) {
    ushort_t* Cb = (ushort_t*)C + sC * batch;
#pragma unroll
    for (int m = 0; m < 4; m++)
#pragma unroll
      for (int r = 0; r < 8; r++) {
        int row = mb + m * 16 + r + lh * 8;
        Cb[row * 256 + col] = f2bf(acc[m][r]);
      }
  } else {
    float* Cb = (float*)C + sC * batch;
#pragma unroll
    for (int m = 0; m < 4; m++)
#pragma unroll
      for (int r = 0; r < 8; r++) {
        int row = mb + m * 16 + r + lh * 8;
        Cb[row * 256 + col] = acc[m][r];
      }
  }
}

// ---------- implicit-GEMM 3x3 conv (one image, NCHW, SAME), bf16 in, WMMA ----------
// outMode: 0 = f32, 1 = bf16, 2 = bf16 + relu
__global__ __launch_bounds__(256) void k_conv3x3(const ushort_t* X, const ushort_t* W,
                                                 const float* bias, void* Y,
                                                 int Cin, int Cout, int outMode) {
  __shared__ __align__(16) ushort_t As[64 * LDA];
  __shared__ __align__(16) ushort_t Bs[128 * LDA];
  const int tid = threadIdx.x;
  const int wave = tid >> 5, lane = tid & 31;
  const int nb = blockIdx.x * 128;           // 128 spatial positions
  const int lh = (lane < 16) ? 0 : 1;
  const int ln = lane & 15;
  const int ksteps = (Cin + 31) >> 5;

  v8f acc[4];
#pragma unroll
  for (int m = 0; m < 4; m++)
#pragma unroll
    for (int r = 0; r < 8; r++) acc[m][r] = 0.0f;

  for (int tap = 0; tap < 9; ++tap) {
    int dy = tap / 3 - 1, dx = tap % 3 - 1;
    for (int ks = 0; ks < ksteps; ++ks) {
      int kc = ks * 32;
#pragma unroll
      for (int i = 0; i < 8; i++) {          // weights tile 64x32 (strided gather)
        int e = tid * 8 + i; int m = e >> 5, k = e & 31;
        int ci = kc + k;
        ushort_t v = 0;
        if (ci < Cin && m < Cout) v = W[(m * Cin + ci) * 9 + tap];
        As[m * LDA + k] = v;
      }
#pragma unroll
      for (int i = 0; i < 16; i++) {         // activation tile 32x128 -> Bs[n][k]
        int e = tid * 16 + i; int k = e >> 7, n = e & 127;
        int ci = kc + k;
        int p = nb + n; int y = p >> 8, x = p & 255;
        int sy = y + dy, sx = x + dx;
        ushort_t v = 0;
        if (ci < Cin && sy >= 0 && sy < 256 && sx >= 0 && sx < 256)
          v = X[(size_t)ci * IMG + sy * 256 + sx];
        Bs[n * LDA + k] = v;
      }
      __syncthreads();
      FragAB fb;
      {
        int c = wave * 16 + ln;
        int kb = lh ? 8 : 0;
        fb.q[0] = *(const uint4*)&Bs[c * LDA + kb];
        fb.q[1] = *(const uint4*)&Bs[c * LDA + kb + 16];
      }
#pragma unroll
      for (int m = 0; m < 4; m++) {
        FragAB fa;
        int row = m * 16 + ln;
        int ka = lh ? 8 : 0;
        fa.q[0] = *(const uint4*)&As[row * LDA + ka];
        fa.q[1] = *(const uint4*)&As[row * LDA + ka + 16];
        acc[m] = __builtin_amdgcn_wmma_f32_16x16x32_bf16(false, fa.v, false, fb.v,
                                                         (short)0, acc[m], false, false);
      }
      __syncthreads();
    }
  }
  int col = nb + wave * 16 + ln;
#pragma unroll
  for (int m = 0; m < 4; m++)
#pragma unroll
    for (int r = 0; r < 8; r++) {
      int row = m * 16 + r + lh * 8;
      if (row < Cout) {
        float vv = acc[m][r] + bias[row];
        if (outMode == 0) {
          ((float*)Y)[(size_t)row * IMG + col] = vv;
        } else {
          if (outMode == 2) vv = fmaxf(vv, 0.0f);
          ((ushort_t*)Y)[(size_t)row * IMG + col] = f2bf(vv);
        }
      }
    }
}

// ---------- dyn MSE sum: (x[:,1:] - hf)^2 ----------
__global__ void k_msedyn(const float* xin, const float* h, float* acc) {
  __shared__ float s[256];
  int tid = threadIdx.x;
  int i = blockIdx.x * 256 + tid;
  int f = i >> 16, p = i & 65535;
  int b = f / 14, t = f % 14;
  float xv = xin[(size_t)(b * 15 + t + 1) * IMG + p];
  float d = xv - h[i];
  s[tid] = d * d;
  __syncthreads();
  for (int o = 128; o > 0; o >>= 1) { if (tid < o) s[tid] += s[tid + o]; __syncthreads(); }
  if (tid == 0) atomicAdd(acc, s[0]);
}

// ---------- generic MSE sum ----------
__global__ void k_mse(const float* a, const float* b, int n, float* acc) {
  __shared__ float s[256];
  int tid = threadIdx.x;
  int i = blockIdx.x * 256 + tid;
  float d = 0.0f;
  if (i < n) { float e = a[i] - b[i]; d = e * e; }
  s[tid] = d;
  __syncthreads();
  for (int o = 128; o > 0; o >>= 1) { if (tid < o) s[tid] += s[tid + o]; __syncthreads(); }
  if (tid == 0) atomicAdd(acc, s[0]);
}

// ---------- finalize ----------
__global__ void k_final(const float* consts, float* out) {
  out[0] = consts[8] / (56.0f * 65536.0f) + consts[9] / (60.0f * 65536.0f);
}

// ---------- host launcher ----------
extern "C" void kernel_launch(void* const* d_in, const int* in_sizes, int n_in,
                              void* d_out, int out_size, void* d_ws, size_t ws_size,
                              hipStream_t stream) {
  const float* xin   = (const float*)d_in[0];
  const float* w_in  = (const float*)d_in[1];  const float* b_in  = (const float*)d_in[2];
  const float* w_hid = (const float*)d_in[3];  const float* b_hid = (const float*)d_in[4];
  const float* w_b1  = (const float*)d_in[5];  const float* b_b1  = (const float*)d_in[6];
  const float* w_b21 = (const float*)d_in[7];  const float* b_b21 = (const float*)d_in[8];
  const float* w_b22 = (const float*)d_in[9];  const float* b_b22 = (const float*)d_in[10];
  const float* w_out = (const float*)d_in[11]; const float* b_out = (const float*)d_in[12];
  float* out = (float*)d_out;

  char* ws = (char*)d_ws;
  size_t o = 0;
  auto take = [&](size_t bytes) { size_t r = o; o = (o + bytes + 255) & ~(size_t)255; return r; };
  float*    consts = (float*)(ws + take(256));
  ushort_t* Sb     = (ushort_t*)(ws + take((size_t)IMG * 2));
  float*    Hinv   = (float*)(ws + take((size_t)IMG * 4));
  size_t FB = (size_t)NF * IMG * 4;
  float*    h    = (float*)(ws + take(FB));
  float*    q    = (float*)(ws + take(FB));
  float*    q1   = (float*)(ws + take(FB));
  float*    qb   = (float*)(ws + take(FB));
  float*    t0f  = (float*)(ws + take(FB));
  ushort_t* t0b  = (ushort_t*)(ws + take((size_t)NF * IMG * 2));
  ushort_t* t1b  = (ushort_t*)(ws + take((size_t)NF * IMG * 2));
  ushort_t* xib  = (ushort_t*)(ws + take((size_t)15 * IMG * 2));
  ushort_t* zb   = (ushort_t*)(ws + take((size_t)64 * IMG * 2));
  ushort_t* z2b  = (ushort_t*)(ws + take((size_t)64 * IMG * 2));
  ushort_t* catb = (ushort_t*)(ws + take((size_t)128 * IMG * 2));
  float*    t64a = (float*)(ws + take((size_t)64 * IMG * 4));
  float*    t64b = (float*)(ws + take((size_t)64 * IMG * 4));
  float*    xa   = (float*)(ws + take((size_t)15 * IMG * 4));
  ushort_t* winb  = (ushort_t*)(ws + take((size_t)64 * 15 * 9 * 2));
  ushort_t* whidb = (ushort_t*)(ws + take((size_t)64 * 64 * 9 * 2));
  ushort_t* wb1b  = (ushort_t*)(ws + take((size_t)64 * 64 * 9 * 2));
  ushort_t* wb21b = (ushort_t*)(ws + take((size_t)64 * 64 * 9 * 2));
  ushort_t* wb22b = (ushort_t*)(ws + take((size_t)64 * 64 * 9 * 2));
  ushort_t* woutb = (ushort_t*)(ws + take((size_t)15 * 128 * 9 * 2));

  k_init0<<<1, 256, 0, stream>>>(consts);
  k_tables<<<256, 256, 0, stream>>>(consts, Sb, Hinv);

  const int NTOT = NF * IMG;
  const int nblk = NTOT / 256;
  k_copyh0<<<nblk, 256, 0, stream>>>(xin, h);
  k_h2pv<<<nblk, 256, 0, stream>>>(xin, consts, qb, q);

  float* qc = q; float* qn = q1;
  dim3 gg(8, NF);
  for (int s = 0; s < 4; ++s) {
    k_qstep<<<nblk, 256, 0, stream>>>(h, qc, qn, consts);
    k_makeqin<<<nblk, 256, 0, stream>>>(qn, qb, t0b);
    k_gemm256<<<gg, 256, 0, stream>>>(Sb, 0, t0b, IMG, t1b, IMG, 1);   // S * qin  -> bf16
    k_gemm256<<<gg, 256, 0, stream>>>(t1b, IMG, Sb, 0, t0f, IMG, 0);   // (.) * S  -> f32
    k_helmcvt<<<nblk, 256, 0, stream>>>(t0f, Hinv, t0b);               // /HELM    -> bf16
    k_gemm256<<<gg, 256, 0, stream>>>(Sb, 0, t0b, IMG, t1b, IMG, 1);   // S * (.)  -> bf16
    k_gemm256<<<gg, 256, 0, stream>>>(t1b, IMG, Sb, 0, t0f, IMG, 0);   // (.) * S  -> f32
    k_finish<<<nblk, 256, 0, stream>>>(t0f, xin, h);
    float* tp = qc; qc = qn; qn = tp;
  }
  k_msedyn<<<nblk, 256, 0, stream>>>(xin, h, &consts[8]);

  // stage conv weights as bf16 (once)
  k_cvt<<<(64 * 15 * 9 + 255) / 256, 256, 0, stream>>>(w_in,  winb,  64 * 15 * 9);
  k_cvt<<<(64 * 64 * 9 + 255) / 256, 256, 0, stream>>>(w_hid, whidb, 64 * 64 * 9);
  k_cvt<<<(64 * 64 * 9 + 255) / 256, 256, 0, stream>>>(w_b1,  wb1b,  64 * 64 * 9);
  k_cvt<<<(64 * 64 * 9 + 255) / 256, 256, 0, stream>>>(w_b21, wb21b, 64 * 64 * 9);
  k_cvt<<<(64 * 64 * 9 + 255) / 256, 256, 0, stream>>>(w_b22, wb22b, 64 * 64 * 9);
  k_cvt<<<(15 * 128 * 9 + 255) / 256, 256, 0, stream>>>(w_out, woutb, 15 * 128 * 9);

  for (int b = 0; b < 4; ++b) {
    const float* xi = xin + (size_t)b * 15 * IMG;
    k_cvt<<<(15 * IMG) / 256, 256, 0, stream>>>(xi, xib, 15 * IMG);
    k_conv3x3<<<512, 256, 0, stream>>>(xib,  winb,  b_in,  zb,   15,  64, 2); // +relu
    k_conv3x3<<<512, 256, 0, stream>>>(zb,   whidb, b_hid, z2b,  64,  64, 1);
    k_conv3x3<<<512, 256, 0, stream>>>(z2b,  wb1b,  b_b1,  catb, 64,  64, 1);
    k_conv3x3<<<512, 256, 0, stream>>>(z2b,  wb21b, b_b21, t64a, 64,  64, 0);
    k_conv3x3<<<512, 256, 0, stream>>>(z2b,  wb22b, b_b22, t64b, 64,  64, 0);
    k_mulcvt<<<(64 * IMG) / 256, 256, 0, stream>>>(t64a, t64b, catb + (size_t)64 * IMG, 64 * IMG);
    k_conv3x3<<<512, 256, 0, stream>>>(catb, woutb, b_out, xa,  128,  15, 0);
    k_mse<<<(15 * IMG) / 256, 256, 0, stream>>>(xi, xa, 15 * IMG, &consts[9]);
  }
  k_final<<<1, 1, 0, stream>>>(consts, out);
}
